// GraphConvLayer_56684978372719
// MI455X (gfx1250) — matrix operand order repaired and checked by reading, not compiled, not run on hardware
//
#include <hip/hip_runtime.h>
#include <hip/hip_bf16.h>

typedef __attribute__((ext_vector_type(16))) _Float16 v16h;
typedef __attribute__((ext_vector_type(8)))  float    v8f;

#define N_NODES 50000
#define N_EDGES 800000
#define FEAT    64
#define BN_EPS  1e-5f

// ---------------------------------------------------------------------------
// Kernel 1: zero the aggregation buffer + BN stat accumulators in workspace
// ---------------------------------------------------------------------------
__global__ void gcn_zero_kernel(float* __restrict__ p, int n) {
    int i = blockIdx.x * blockDim.x + threadIdx.x;
    if (i < n) p[i] = 0.0f;
}

// ---------------------------------------------------------------------------
// Kernel 2: edge scatter-add of RAW features (linearity lets us aggregate
// before the linear transform).  64 threads per edge -> coalesced feature row
// read, coalesced atomic row update.  feature table (12.8 MB) lives in L2.
// ---------------------------------------------------------------------------
__global__ __launch_bounds__(256)
void gcn_scatter_kernel(const float* __restrict__ feat,
                        const int*   __restrict__ src,
                        const int*   __restrict__ dst,
                        float*       __restrict__ agg) {
    long long tid = (long long)blockIdx.x * blockDim.x + threadIdx.x;
    int e = (int)(tid >> 6);      // edge index
    int f = (int)(tid & 63);      // feature index
    if (e < N_EDGES) {
        int s = src[e];
        int d = dst[e];
        atomicAdd(&agg[(long long)d * FEAT + f], feat[(long long)s * FEAT + f]);
    }
}

// ---------------------------------------------------------------------------
// Kernel 3: h = agg @ W^T via v_wmma_f32_16x16x32_f16, fused BN sum/sumsq.
// One wave32 per 16-row M-tile; K=64 as two K=32 WMMA steps; 4 N-tiles.
// ---------------------------------------------------------------------------
__global__ __launch_bounds__(256)
void gcn_wmma_gemm_kernel(const float* __restrict__ agg,
                          const float* __restrict__ W,      // [64 out][64 in] row-major
                          float*       __restrict__ h,      // [N_NODES][64] -> d_out
                          float*       __restrict__ sum,
                          float*       __restrict__ sumsq) {
    const int wave = (blockIdx.x * blockDim.x + threadIdx.x) >> 5;   // wave-uniform
    if (wave >= N_NODES / 16) return;                                // uniform exit: EXEC stays all-1s
    const int lane = threadIdx.x & 31;
    const int grp  = lane >> 4;     // 0: lanes 0-15, 1: lanes 16-31
    const int l15  = lane & 15;

    // ---- A fragments (16x32 f16, per documented layout) for k0 = 0 and 32 ----
    const int m = wave * 16 + l15;
    const float* rp = agg + (long long)m * FEAT;
    v16h a[2];
#pragma unroll
    for (int kb = 0; kb < 2; ++kb) {
        const int k0 = kb * 32;
        const float4* q0 = (const float4*)(rp + k0 + 8 * grp);       // halves 0..7
        const float4* q1 = (const float4*)(rp + k0 + 16 + 8 * grp);  // halves 8..15
        float4 x0 = q0[0], x1 = q0[1], x2 = q1[0], x3 = q1[1];
        a[kb][0] = (_Float16)x0.x; a[kb][1] = (_Float16)x0.y;
        a[kb][2] = (_Float16)x0.z; a[kb][3] = (_Float16)x0.w;
        a[kb][4] = (_Float16)x1.x; a[kb][5] = (_Float16)x1.y;
        a[kb][6] = (_Float16)x1.z; a[kb][7] = (_Float16)x1.w;
        a[kb][8]  = (_Float16)x2.x; a[kb][9]  = (_Float16)x2.y;
        a[kb][10] = (_Float16)x2.z; a[kb][11] = (_Float16)x2.w;
        a[kb][12] = (_Float16)x3.x; a[kb][13] = (_Float16)x3.y;
        a[kb][14] = (_Float16)x3.z; a[kb][15] = (_Float16)x3.w;
    }

    // ---- loop over the 4 N-tiles of the 64 output features ----
#pragma unroll
    for (int t = 0; t < 4; ++t) {
        const int n = t * 16 + l15;            // output feature this lane owns (B col / D col)
        const float* wr = W + (long long)n * FEAT;   // row n of W == column n of W^T
        v8f c = {};
#pragma unroll
        for (int kb = 0; kb < 2; ++kb) {
            const int k0 = kb * 32;
            // B fragment (32x16 f16): lane holds K = 8*grp+0..7 and 16+8*grp+0..7 of col n
            const float4* qb0 = (const float4*)(wr + k0 + 8 * grp);
            const float4* qb1 = (const float4*)(wr + k0 + 16 + 8 * grp);
            float4 b0 = qb0[0], b1 = qb0[1], b2 = qb1[0], b3 = qb1[1];
            v16h b;
            b[0]  = (_Float16)b0.x; b[1]  = (_Float16)b0.y;
            b[2]  = (_Float16)b0.z; b[3]  = (_Float16)b0.w;
            b[4]  = (_Float16)b1.x; b[5]  = (_Float16)b1.y;
            b[6]  = (_Float16)b1.z; b[7]  = (_Float16)b1.w;
            b[8]  = (_Float16)b2.x; b[9]  = (_Float16)b2.y;
            b[10] = (_Float16)b2.z; b[11] = (_Float16)b2.w;
            b[12] = (_Float16)b3.x; b[13] = (_Float16)b3.y;
            b[14] = (_Float16)b3.z; b[15] = (_Float16)b3.w;
            c = __builtin_amdgcn_wmma_f32_16x16x32_f16(
                    /*neg_a=*/false, a[kb], /*neg_b=*/false, b,
                    /*c_mod=*/(short)0, c, /*reuse_a=*/false, /*reuse_b=*/false);
        }
        // ---- write D tile + accumulate BN statistics ----
        float s = 0.0f, s2 = 0.0f;
#pragma unroll
        for (int j = 0; j < 8; ++j) {
            const int row = wave * 16 + j + 8 * grp;   // D layout: VGPR j -> M=j (+8 for hi lanes)
            const float v = c[j];
            h[(long long)row * FEAT + n] = v;
            s  += v;
            s2 += v * v;
        }
        atomicAdd(&sum[n],   s);
        atomicAdd(&sumsq[n], s2);
    }
}

// ---------------------------------------------------------------------------
// Kernel 4: BatchNorm (biased batch stats) + affine + ReLU, in place on d_out
// ---------------------------------------------------------------------------
__global__ __launch_bounds__(256)
void gcn_bn_relu_kernel(float* __restrict__ h,
                        const float* __restrict__ sum,
                        const float* __restrict__ sumsq,
                        const float* __restrict__ gamma,
                        const float* __restrict__ beta) {
    int i = blockIdx.x * blockDim.x + threadIdx.x;
    if (i >= N_NODES * FEAT) return;
    const int f = i & 63;
    const float invN = 1.0f / (float)N_NODES;
    const float mean = sum[f] * invN;
    const float var  = sumsq[f] * invN - mean * mean;
    const float inv  = rsqrtf(var + BN_EPS);
    float v = (h[i] - mean) * inv * gamma[f] + beta[f];
    h[i] = v > 0.0f ? v : 0.0f;
}

// ---------------------------------------------------------------------------
extern "C" void kernel_launch(void* const* d_in, const int* in_sizes, int n_in,
                              void* d_out, int out_size, void* d_ws, size_t ws_size,
                              hipStream_t stream) {
    const float* feature = (const float*)d_in[0];
    const float* W       = (const float*)d_in[1];
    const float* gamma   = (const float*)d_in[2];
    const float* beta    = (const float*)d_in[3];
    const int*   src     = (const int*)d_in[4];
    const int*   dst     = (const int*)d_in[5];
    float* out = (float*)d_out;

    // workspace layout: [agg: N_NODES*FEAT][sum: 64][sumsq: 64]
    float* agg   = (float*)d_ws;
    float* sum   = agg + (long long)N_NODES * FEAT;
    float* sumsq = sum + FEAT;

    const int zero_n = N_NODES * FEAT + 2 * FEAT;
    gcn_zero_kernel<<<(zero_n + 255) / 256, 256, 0, stream>>>(agg, zero_n);

    const long long scatter_threads = (long long)N_EDGES * FEAT;
    gcn_scatter_kernel<<<(int)((scatter_threads + 255) / 256), 256, 0, stream>>>(
        feature, src, dst, agg);

    const int n_waves  = N_NODES / 16;               // 3125 M-tiles, 1 wave each
    const int n_blocks = (n_waves + 7) / 8;          // 8 waves per 256-thread block
    gcn_wmma_gemm_kernel<<<n_blocks, 256, 0, stream>>>(agg, W, out, sum, sumsq);

    gcn_bn_relu_kernel<<<(N_NODES * FEAT + 255) / 256, 256, 0, stream>>>(
        out, sum, sumsq, gamma, beta);
}
